// WholeBrainRateModel_11725260718115
// MI455X (gfx1250) — matrix operand order repaired and checked by reading, not compiled, not run on hardware
//
#include <hip/hip_runtime.h>
#include <hip/hip_bf16.h>

// Model dims (fixed by reference)
#define B_   2
#define N_   50000
#define H_   64
#define E_   1000000
#define OBS_ 128
#define A_   18
#define AFF_ 512
#define EFF_ 256

typedef __attribute__((ext_vector_type(16))) __bf16 v16bf;
typedef __attribute__((ext_vector_type(8)))  float  v8f;

__device__ __forceinline__ v8f wmma_bf16(v16bf a, v16bf b, v8f c) {
  // D = A(16x32 bf16) * B(32x16 bf16) + C(16x16 f32)
  return __builtin_amdgcn_wmma_f32_16x16x32_bf16(false, a, false, b, (short)0, c, false, false);
}

// ---------------------------------------------------------------- zero
__global__ __launch_bounds__(256) void k_zero(float* __restrict__ msgs,
                                              int* __restrict__ mask) {
  size_t i = (size_t)blockIdx.x * blockDim.x + threadIdx.x;
  size_t stride = (size_t)gridDim.x * blockDim.x;
  const size_t nmsg = (size_t)B_ * N_ * H_;
  for (size_t j = i; j < nmsg; j += stride) msgs[j] = 0.0f;
  for (size_t j = i; j < (size_t)N_; j += stride) mask[j] = 0;
}

// ---------------------------------------------------------------- afferent mask
__global__ __launch_bounds__(256) void k_mask(const int* __restrict__ aff,
                                              int* __restrict__ mask) {
  int i = blockIdx.x * blockDim.x + threadIdx.x;
  if (i < AFF_) mask[aff[i]] = 1;
}

// ---------------------------------------------------------------- obs projection
__global__ __launch_bounds__(128) void k_proj(const float* __restrict__ obs,
                                              const float* __restrict__ W_in,
                                              const float* __restrict__ b_in,
                                              float* __restrict__ projected) {
  int tid = threadIdx.x;           // 128 threads: (b, h)
  int b = tid >> 6, h = tid & 63;
  float s = b_in[h];
  for (int k = 0; k < OBS_; ++k) s += obs[b * OBS_ + k] * W_in[k * H_ + h];
  projected[b * H_ + h] = s;
}

// ---------------------------------------------------------------- pack weights into WMMA B-operand layout
// B(32x16) layout: lane l holds column n = l%16; element e <-> K = (l/16)*8 + (e&7) + (e>>3)*16.
// Wmsg packed:  [t(4)][kh(2)][lane(32)][e(16)]  (K = kh*32 + ...)
// Wgate/Wcand:  [t(4)][s(4)][lane(32)][e(16)]   (K = s*32 + ...)
__global__ __launch_bounds__(256) void k_pack(const float* __restrict__ Wmsg,
                                              const float* __restrict__ Wgate,
                                              const float* __restrict__ Wcand,
                                              __bf16* __restrict__ Wmp,
                                              __bf16* __restrict__ Wgp,
                                              __bf16* __restrict__ Wcp) {
  int tid = blockIdx.x * blockDim.x + threadIdx.x;
  int stride = gridDim.x * blockDim.x;
  for (int i = tid; i < 4 * 2 * 32 * 16; i += stride) {
    int e = i & 15, lane = (i >> 4) & 31, th = i >> 9;
    int t = th >> 1, kh = th & 1;
    int k = kh * 32 + (lane >> 4) * 8 + (e & 7) + (e >> 3) * 16;
    int n = t * 16 + (lane & 15);
    Wmp[i] = (__bf16)Wmsg[k * H_ + n];
  }
  for (int i = tid; i < 4 * 4 * 32 * 16; i += stride) {
    int e = i & 15, lane = (i >> 4) & 31, ts = i >> 9;
    int t = ts >> 2, s = ts & 3;
    int k = s * 32 + (lane >> 4) * 8 + (e & 7) + (e >> 3) * 16;
    int n = t * 16 + (lane & 15);
    Wgp[i] = (__bf16)Wgate[k * H_ + n];
    Wcp[i] = (__bf16)Wcand[k * H_ + n];
  }
}

// ---------------------------------------------------------------- edge message pass (WMMA + atomic scatter)
// One wave handles a tile of 16 edges x full H=64 output (4 N-tiles, K=64 -> 2 chained WMMAs each).
__global__ __launch_bounds__(256) void k_edges(const float* __restrict__ state,
                                               const __bf16* __restrict__ Wmp,
                                               const int* __restrict__ src,
                                               const int* __restrict__ dst,
                                               float* __restrict__ messages) {
  const int lane = threadIdx.x & 31;
  const int g = lane >> 4;            // lane group (0/1)
  const int ln = lane & 15;
  const int wave = blockIdx.x * (blockDim.x >> 5) + (threadIdx.x >> 5);
  const int nwaves = gridDim.x * (blockDim.x >> 5);
  const int ntiles = E_ / 16;         // 62500, exact

  // W_msg B operands resident in VGPRs: [t][khalf]
  v16bf Bv[4][2];
#pragma unroll
  for (int t = 0; t < 4; ++t)
#pragma unroll
    for (int kh = 0; kh < 2; ++kh)
      Bv[t][kh] = *(const v16bf*)(Wmp + (((t * 2 + kh) * 32) + lane) * 16);

  for (int tile = wave; tile < ntiles; tile += nwaves) {
    const int ebase = tile * 16;
    const int arow = src[ebase + ln];      // A row for this lane (M = ln)
    int drows[8];
#pragma unroll
    for (int j = 0; j < 8; ++j) drows[j] = dst[ebase + j + 8 * g];   // M = j + 8g

#pragma unroll
    for (int b = 0; b < B_; ++b) {
      const float* rp = state + ((size_t)b * N_ + arow) * H_;
      // Build two A operands (K 0..31 and 32..63), bf16-converted on the fly.
      v16bf A0, A1;
#pragma unroll
      for (int half = 0; half < 2; ++half) {
        const float* q0 = rp + half * 32 + g * 8;        // K chunk 1
        const float* q1 = rp + half * 32 + 16 + g * 8;   // K chunk 2
        float4 c0 = *(const float4*)q0;
        float4 c1 = *(const float4*)(q0 + 4);
        float4 c2 = *(const float4*)q1;
        float4 c3 = *(const float4*)(q1 + 4);
        v16bf& A = half ? A1 : A0;
        A[0] = (__bf16)c0.x;  A[1] = (__bf16)c0.y;  A[2] = (__bf16)c0.z;  A[3] = (__bf16)c0.w;
        A[4] = (__bf16)c1.x;  A[5] = (__bf16)c1.y;  A[6] = (__bf16)c1.z;  A[7] = (__bf16)c1.w;
        A[8] = (__bf16)c2.x;  A[9] = (__bf16)c2.y;  A[10] = (__bf16)c2.z; A[11] = (__bf16)c2.w;
        A[12] = (__bf16)c3.x; A[13] = (__bf16)c3.y; A[14] = (__bf16)c3.z; A[15] = (__bf16)c3.w;
      }

      float* mb = messages + (size_t)b * N_ * H_;
#pragma unroll
      for (int t = 0; t < 4; ++t) {
        v8f acc = {};
        acc = wmma_bf16(A0, Bv[t][0], acc);
        acc = wmma_bf16(A1, Bv[t][1], acc);
        const int col = t * 16 + ln;        // N = ln
#pragma unroll
        for (int j = 0; j < 8; ++j)
          atomicAdd(mb + (size_t)drows[j] * H_ + col, acc[j]);
      }
    }
  }
}

// ---------------------------------------------------------------- GRU update (WMMA)
// One wave handles 16 nodes of one batch: gate_in = [state | messages+inj] (K=128),
// z/cand via 2x(4 tiles x 4 K-steps) WMMAs, then blend and write next_state.
__global__ __launch_bounds__(256) void k_gru(const float* __restrict__ state,
                                             const float* __restrict__ messages,
                                             const float* __restrict__ projected,
                                             const int* __restrict__ mask,
                                             const __bf16* __restrict__ Wgp,
                                             const __bf16* __restrict__ Wcp,
                                             const float* __restrict__ b_gate,
                                             const float* __restrict__ b_cand,
                                             float* __restrict__ next_state) {
  const int lane = threadIdx.x & 31;
  const int g = lane >> 4;
  const int ln = lane & 15;
  const int wave = blockIdx.x * (blockDim.x >> 5) + (threadIdx.x >> 5);
  const int nwaves = gridDim.x * (blockDim.x >> 5);
  const int tiles_per_b = N_ / 16;    // 3125, exact
  const int ntasks = B_ * tiles_per_b;

  for (int task = wave; task < ntasks; task += nwaves) {
    const int b = task / tiles_per_b;
    const int n0 = (task % tiles_per_b) * 16;
    const int arow = n0 + ln;                     // A row M = ln
    const float* srow = state + ((size_t)b * N_ + arow) * H_;
    const float* mrow = messages + ((size_t)b * N_ + arow) * H_;
    const float injf = mask[arow] ? 1.0f : 0.0f;
    const float* pj = projected + b * H_;

    v16bf A[4];   // K-steps of 32 over K=128
#pragma unroll
    for (int s = 0; s < 4; ++s) {
      const int kb = s * 32 + g * 8;
#pragma unroll
      for (int i = 0; i < 8; ++i) {
        const int k0 = kb + i;
        const int k1 = kb + 16 + i;
        float x0 = (k0 < H_) ? srow[k0] : (mrow[k0 - H_] + injf * pj[k0 - H_]);
        float x1 = (k1 < H_) ? srow[k1] : (mrow[k1 - H_] + injf * pj[k1 - H_]);
        A[s][i] = (__bf16)x0;
        A[s][8 + i] = (__bf16)x1;
      }
    }

#pragma unroll
    for (int t = 0; t < 4; ++t) {
      v8f za = {}, ca = {};
#pragma unroll
      for (int s = 0; s < 4; ++s) {
        v16bf Bg = *(const v16bf*)(Wgp + (((t * 4 + s) * 32) + lane) * 16);
        za = wmma_bf16(A[s], Bg, za);
      }
#pragma unroll
      for (int s = 0; s < 4; ++s) {
        v16bf Bc = *(const v16bf*)(Wcp + (((t * 4 + s) * 32) + lane) * 16);
        ca = wmma_bf16(A[s], Bc, ca);
      }
      const int col = t * 16 + ln;
      const float bg = b_gate[col];
      const float bc = b_cand[col];
#pragma unroll
      for (int j = 0; j < 8; ++j) {
        const int node = n0 + j + 8 * g;          // M = j + 8g
        const size_t idx = ((size_t)b * N_ + node) * H_ + col;
        float z = 1.0f / (1.0f + __expf(-(za[j] + bg)));
        float c = tanhf(ca[j] + bc);
        float so = state[idx];
        next_state[idx] = (1.0f - z) * so + z * c;
      }
    }
  }
}

// ---------------------------------------------------------------- readout + policy heads (tiny)
__global__ __launch_bounds__(256) void k_head(const float* __restrict__ next_state,
                                              const int* __restrict__ eff,
                                              const float* __restrict__ W_dec,
                                              const float* __restrict__ b_dec,
                                              const float* __restrict__ W_mean,
                                              const float* __restrict__ b_mean,
                                              const float* __restrict__ W_ls,
                                              const float* __restrict__ b_ls,
                                              float* __restrict__ out) {
  __shared__ float readout[B_][H_];
  __shared__ float decoded[B_][H_];
  const int tid = threadIdx.x;
  if (tid < B_ * H_) {
    int b = tid >> 6, h = tid & 63;
    float s = 0.0f;
    for (int i = 0; i < EFF_; ++i)
      s += next_state[((size_t)b * N_ + eff[i]) * H_ + h];
    readout[b][h] = s * (1.0f / (float)EFF_);
  }
  __syncthreads();
  if (tid < B_ * H_) {
    int b = tid >> 6, h = tid & 63;
    float s = b_dec[h];
    for (int k = 0; k < H_; ++k) s += readout[b][k] * W_dec[k * H_ + h];
    decoded[b][h] = tanhf(s);
  }
  __syncthreads();
  if (tid < B_ * A_) {                       // mean
    int b = tid / A_, a = tid % A_;
    float s = b_mean[a];
    for (int k = 0; k < H_; ++k) s += decoded[b][k] * W_mean[k * A_ + a];
    out[tid] = s;
  } else if (tid < 2 * B_ * A_) {            // log_std
    int t2 = tid - B_ * A_;
    int b = t2 / A_, a = t2 % A_;
    float s = b_ls[a];
    for (int k = 0; k < H_; ++k) s += decoded[b][k] * W_ls[k * A_ + a];
    s = fminf(fmaxf(s, -5.0f), 2.0f);
    out[B_ * A_ + t2] = s;
  }
}

// ---------------------------------------------------------------- launch
extern "C" void kernel_launch(void* const* d_in, const int* in_sizes, int n_in,
                              void* d_out, int out_size, void* d_ws, size_t ws_size,
                              hipStream_t stream) {
  const float* obs    = (const float*)d_in[0];
  const float* state  = (const float*)d_in[1];
  const float* W_in   = (const float*)d_in[2];
  const float* b_in   = (const float*)d_in[3];
  const float* W_msg  = (const float*)d_in[4];
  const float* W_gate = (const float*)d_in[5];
  const float* b_gate = (const float*)d_in[6];
  const float* W_cand = (const float*)d_in[7];
  const float* b_cand = (const float*)d_in[8];
  const float* W_dec  = (const float*)d_in[9];
  const float* b_dec  = (const float*)d_in[10];
  const float* W_mean = (const float*)d_in[11];
  const float* b_mean = (const float*)d_in[12];
  const float* W_ls   = (const float*)d_in[13];
  const float* b_ls   = (const float*)d_in[14];
  const int* src      = (const int*)d_in[15];
  const int* dst      = (const int*)d_in[16];
  const int* aff      = (const int*)d_in[17];
  const int* eff      = (const int*)d_in[18];
  float* out = (float*)d_out;

  // Workspace carve-up (all offsets 32B aligned)
  float*  msgs      = (float*)d_ws;                          // 6,400,000 f32 (25.6 MB)
  float*  projected = msgs + (size_t)B_ * N_ * H_;           // 128 f32
  int*    mask      = (int*)(projected + B_ * H_);           // 50,000 i32
  __bf16* Wmp       = (__bf16*)(mask + N_);                  // 4096 bf16
  __bf16* Wgp       = Wmp + 4 * 2 * 32 * 16;                 // 8192 bf16
  __bf16* Wcp       = Wgp + 4 * 4 * 32 * 16;                 // 8192 bf16

  float* next_state = out + 2 * B_ * A_;                     // d_out offset 72

  k_zero<<<2048, 256, 0, stream>>>(msgs, mask);
  k_mask<<<2, 256, 0, stream>>>(aff, mask);
  k_proj<<<1, 128, 0, stream>>>(obs, W_in, b_in, projected);
  k_pack<<<32, 256, 0, stream>>>(W_msg, W_gate, W_cand, Wmp, Wgp, Wcp);
  k_edges<<<1024, 256, 0, stream>>>(state, Wmp, src, dst, msgs);
  k_gru<<<782, 256, 0, stream>>>(state, msgs, projected, mask, Wgp, Wcp,
                                 b_gate, b_cand, next_state);
  k_head<<<1, 256, 0, stream>>>(next_state, eff, W_dec, b_dec,
                                W_mean, b_mean, W_ls, b_ls, out);
}
